// CausalSelfAttention_33088428048914
// MI455X (gfx1250) — matrix operand order repaired and checked by reading, not compiled, not run on hardware
//
#include <hip/hip_runtime.h>
#include <hip/hip_bf16.h>
#include <math.h>

// ---------------------------------------------------------------------------
// CausalSelfAttention for MI455X (gfx1250), wave32 + WMMA bf16 path.
//   x[B,T,C] fp32 -> bf16 ; qkv = x@Wqkv^T + b (WMMA, async-LDS weight tiles)
//   flash attention (WMMA S=QK^T, online softmax, WMMA O=PV)
//   out = y@Wo^T + bo (WMMA, async-LDS weight tiles, fp32 result)
// ---------------------------------------------------------------------------

#define Bq   4
#define Tq   2048
#define Cq   1024
#define Hq   16
#define HDq  64
#define C3q  3072
#define NTOK (Bq * Tq)          // 8192

typedef __bf16 bf16;
typedef __attribute__((ext_vector_type(16))) __bf16 v16bf;
typedef __attribute__((ext_vector_type(8)))  __bf16 v8bf;
typedef __attribute__((ext_vector_type(8)))  float  v8f;

union frag_u { v16bf v; v8bf h[2]; };

// ---- A-fragment: 16x32 bf16, row-major source, row = base + (lane&15) ------
// lanes 0-15 : elems 0..7 = K k0..k0+7,  elems 8..15 = K k0+16..k0+23
// lanes 16-31: elems 0..7 = K k0+8..+15, elems 8..15 = K k0+24..+31
__device__ __forceinline__ v16bf load_a(const bf16* rowptr, int k0, int lane) {
    frag_u u;
    const bf16* p = rowptr + k0;
    u.h[0] = *(const v8bf*)(p + (lane < 16 ? 0  : 8));
    u.h[1] = *(const v8bf*)(p + (lane < 16 ? 16 : 24));
    return u.v;
}

// ---- B-fragment: 32x16 bf16 (K x N), source row-major along K --------------
// lane owns N = lane&15 ; lanes 0-15 hold K k0..+7,k0+16..+23; lanes 16-31 k0+8..+15,k0+24..+31
__device__ __forceinline__ v16bf load_b(const bf16* base, size_t ld,
                                        int nrow, int k0, int lane) {
    frag_u u;
    const bf16* p = base + (size_t)nrow * ld + k0 + (lane < 16 ? 0 : 8);
    u.h[0] = *(const v8bf*)(p);
    u.h[1] = *(const v8bf*)(p + 16);
    return u.v;
}

__device__ __forceinline__ v8f wmma_bf16(v16bf a, v16bf b, v8f c) {
    return __builtin_amdgcn_wmma_f32_16x16x32_bf16(
        false, a, false, b, (short)0, c, false, false);
}

// ---- CDNA5 async global -> LDS copy (16B per lane), ASYNCcnt-tracked -------
__device__ __forceinline__ void async_ld_b128(unsigned ldsoff, const void* gptr) {
    asm volatile("global_load_async_to_lds_b128 %0, %1, off"
                 :: "v"(ldsoff), "v"((unsigned long long)(uintptr_t)gptr)
                 : "memory");
}
__device__ __forceinline__ void wait_async0() {
    asm volatile("s_wait_asynccnt 0x0" ::: "memory");
}
// LDS byte offset of a shared-memory pointer (aperture rule: addr[31:0])
__device__ __forceinline__ unsigned lds_off(const void* p) {
    return (unsigned)(uintptr_t)p;
}

// ---------------------------------------------------------------------------
// fp32 -> bf16 conversion (vectorized)
// ---------------------------------------------------------------------------
__global__ void cvt_f32_bf16(const float* __restrict__ src,
                             bf16* __restrict__ dst, int n) {
    int i = (blockIdx.x * blockDim.x + threadIdx.x) * 4;
    if (i + 3 < n) {
        float4 v = *(const float4*)(src + i);
        dst[i + 0] = (bf16)v.x;
        dst[i + 1] = (bf16)v.y;
        dst[i + 2] = (bf16)v.z;
        dst[i + 3] = (bf16)v.w;
    } else {
        for (; i < n; ++i) dst[i] = (bf16)src[i];
    }
}

// ---------------------------------------------------------------------------
// QKV projection: qkv = X @ Wqkv^T + bqkv, scattered into head-major layouts.
// Block = 8 waves; wave w -> 16(M) x 64(N) tile. Weight tiles (64N x 64K bf16,
// 8KB) are staged global->LDS with async copies, double-buffered.
// grid = (3072/64, 8192/128).
// ---------------------------------------------------------------------------
__global__ __launch_bounds__(256) void qkv_gemm(
    const bf16* __restrict__ X, const bf16* __restrict__ W,
    const float* __restrict__ bias,
    bf16* __restrict__ Qb, bf16* __restrict__ Kb, bf16* __restrict__ Vtb) {

    __shared__ __align__(16) bf16 wtile[2][64 * 64];

    const int tid   = threadIdx.x;
    const int lane  = tid & 31;
    const int wv    = tid >> 5;
    const int mbase = blockIdx.y * 128 + wv * 16;
    const int nbase = blockIdx.x * 64;

    // per-thread async-copy slot: 32B (2 x b128) of the 64x64 bf16 tile
    const int crow = tid >> 2;            // weight row within tile (0..63)
    const int cel  = (tid & 3) * 16;      // element offset within row
    const bf16* wsrc = W + (size_t)(nbase + crow) * Cq + cel;
    const unsigned ldsdst = lds_off(&wtile[0][0]) + (unsigned)(crow * 64 + cel) * 2;

    // prologue: stage K-chunk 0 into buffer 0
    async_ld_b128(ldsdst,      wsrc);
    async_ld_b128(ldsdst + 16, wsrc + 8);

    v8f acc[4] = {v8f{}, v8f{}, v8f{}, v8f{}};
    const bf16* arow = X + (size_t)(mbase + (lane & 15)) * Cq;

    for (int k0 = 0; k0 < Cq; k0 += 64) {
        const unsigned cur = (unsigned)(k0 >> 6) & 1u;
        wait_async0();                    // my stores into wtile[cur] landed
        __syncthreads();                  // whole tile visible; prev buf free
        if (k0 + 64 < Cq) {               // overlap next chunk's DMA
            const bf16* s = wsrc + k0 + 64;
            const unsigned d = ldsdst + (cur ^ 1u) * (64 * 64 * 2);
            async_ld_b128(d,      s);
            async_ld_b128(d + 16, s + 8);
            __builtin_prefetch(arow + k0 + 64, 0, 3);
        }
        const bf16* bt = &wtile[cur][0];
#pragma unroll
        for (int ks = 0; ks < 2; ++ks) {
            // batch all fragment loads, then issue the 4 WMMAs back-to-back
            v16bf a = load_a(arow, k0 + ks * 32, lane);
            v16bf bfr[4];
#pragma unroll
            for (int nt = 0; nt < 4; ++nt)
                bfr[nt] = load_b(bt, 64, nt * 16 + (lane & 15), ks * 32, lane);
#pragma unroll
            for (int nt = 0; nt < 4; ++nt)
                acc[nt] = wmma_bf16(a, bfr[nt], acc[nt]);
        }
    }

    // epilogue: + bias, scatter to Q (scaled), K, V^T
#pragma unroll
    for (int nt = 0; nt < 4; ++nt) {
        const int m   = nbase + nt * 16 + (lane & 15);   // channel in [0,3072)
        const float bv = bias[m];
        const int sec = m >> 10;                         // 0:Q 1:K 2:V
        const int mm  = m & (Cq - 1);
        const int h   = mm >> 6, d = mm & 63;
#pragma unroll
        for (int r = 0; r < 8; ++r) {
            const int row = mbase + r + (lane < 16 ? 0 : 8);   // token
            const int bb  = row >> 11, t = row & (Tq - 1);
            const float v = acc[nt][r] + bv;
            if (sec == 0) {
                Qb[((size_t)(bb * Hq + h) * Tq + t) * HDq + d] = (bf16)(v * 0.125f);
            } else if (sec == 1) {
                Kb[((size_t)(bb * Hq + h) * Tq + t) * HDq + d] = (bf16)v;
            } else {
                Vtb[((size_t)(bb * Hq + h) * HDq + d) * Tq + t] = (bf16)v;
            }
        }
    }
}

// ---------------------------------------------------------------------------
// Flash attention: wave = 16 queries of one (b,h); 32-key blocks, online
// softmax; S and O via WMMA; P re-laid-out through per-wave LDS tile.
// grid = (T/128, H, B), block = 256.
// ---------------------------------------------------------------------------
__global__ __launch_bounds__(256) void attn_kernel(
    const bf16* __restrict__ Qb, const bf16* __restrict__ Kb,
    const bf16* __restrict__ Vtb, const int* __restrict__ amask,
    bf16* __restrict__ Yb) {

    __shared__ __align__(16) bf16 pstage[8][16 * 32];

    const int lane  = threadIdx.x & 31;
    const int wv    = threadIdx.x >> 5;
    const int h     = blockIdx.y;
    const int b     = blockIdx.z;
    const int qbase = (blockIdx.x * 8 + wv) * 16;

    const bf16* Qh = Qb  + (size_t)(b * Hq + h) * Tq * HDq;
    const bf16* Kh = Kb  + (size_t)(b * Hq + h) * Tq * HDq;
    const bf16* Vh = Vtb + (size_t)(b * Hq + h) * HDq * Tq;
    const int*  mr = amask + b * Tq;

    // Q fragments (2 K-steps over hd=64), Q already pre-scaled by 1/sqrt(hd)
    v16bf aq[2];
    const bf16* qrowp = Qh + (size_t)(qbase + (lane & 15)) * HDq;
    aq[0] = load_a(qrowp, 0, lane);
    aq[1] = load_a(qrowp, 32, lane);

    v8f o[4] = {v8f{}, v8f{}, v8f{}, v8f{}};
    float mst[8], lst[8];
#pragma unroll
    for (int r = 0; r < 8; ++r) { mst[r] = -__builtin_inff(); lst[r] = 0.f; }

    const int nkt = (qbase + 15) / 32 + 1;      // causal: keys <= last query
    for (int kt = 0; kt < nkt; ++kt) {
        const int kbase = kt * 32;

        // ---- S = Q @ K^T (16 x 32 scores); batch loads, group WMMAs ----
        v8f s0 = {}, s1 = {};
#pragma unroll
        for (int ks = 0; ks < 2; ++ks) {
            v16bf bk0 = load_b(Kh, HDq, kbase + (lane & 15),      ks * 32, lane);
            v16bf bk1 = load_b(Kh, HDq, kbase + 16 + (lane & 15), ks * 32, lane);
            s0 = wmma_bf16(aq[ks], bk0, s0);
            s1 = wmma_bf16(aq[ks], bk1, s1);
        }

        // ---- causal + key-validity masking ----
        const int t0 = kbase + (lane & 15);
        const int t1 = t0 + 16;
        const bool km0 = (mr[t0] != 0);
        const bool km1 = (mr[t1] != 0);
#pragma unroll
        for (int r = 0; r < 8; ++r) {
            const int qrow = qbase + r + (lane < 16 ? 0 : 8);
            if (t0 > qrow || !km0) s0[r] = -__builtin_inff();
            if (t1 > qrow || !km1) s1[r] = -__builtin_inff();
        }

        // ---- online softmax (row = 16 lanes of one half; xor<=8 stays in half)
        float alpha[8];
#pragma unroll
        for (int r = 0; r < 8; ++r) {
            float mx = fmaxf(s0[r], s1[r]);
            for (int off = 8; off >= 1; off >>= 1)
                mx = fmaxf(mx, __shfl_xor(mx, off, 32));
            const float mnew = fmaxf(mst[r], mx);
            alpha[r] = __expf(mst[r] - mnew);          // 0 on first block
            mst[r]   = mnew;
            const float p0 = __expf(s0[r] - mnew);     // masked -> exp(-inf)=0
            const float p1 = __expf(s1[r] - mnew);
            s0[r] = p0; s1[r] = p1;
            float sum = p0 + p1;
            for (int off = 8; off >= 1; off >>= 1)
                sum += __shfl_xor(sum, off, 32);
            lst[r] = lst[r] * alpha[r] + sum;
            o[0][r] *= alpha[r]; o[1][r] *= alpha[r];
            o[2][r] *= alpha[r]; o[3][r] *= alpha[r];
        }

        // ---- P: C-layout -> A-layout via per-wave LDS tile (bf16) ----
        bf16* ps = &pstage[wv][0];
#pragma unroll
        for (int r = 0; r < 8; ++r) {
            const int m = r + (lane < 16 ? 0 : 8);
            ps[m * 32 + (lane & 15)]      = (bf16)s0[r];
            ps[m * 32 + 16 + (lane & 15)] = (bf16)s1[r];
        }
        const v16bf ap = load_a(ps + (size_t)(lane & 15) * 32, 0, lane);

        // ---- O += P @ V (V stored transposed: row=d, contiguous t) ----
        v16bf bv[4];
#pragma unroll
        for (int nt = 0; nt < 4; ++nt)
            bv[nt] = load_b(Vh, Tq, nt * 16 + (lane & 15), kbase, lane);
#pragma unroll
        for (int nt = 0; nt < 4; ++nt)
            o[nt] = wmma_bf16(ap, bv[nt], o[nt]);
    }

    // ---- normalize, apply query mask, store y as bf16 [B*T, C] ----
#pragma unroll
    for (int r = 0; r < 8; ++r) {
        const int qrow = qbase + r + (lane < 16 ? 0 : 8);
        const float inv = (mr[qrow] != 0 && lst[r] > 0.f) ? (1.f / lst[r]) : 0.f;
        bf16* yrow = Yb + (size_t)(b * Tq + qrow) * Cq + h * HDq;
#pragma unroll
        for (int nt = 0; nt < 4; ++nt)
            yrow[nt * 16 + (lane & 15)] = (bf16)(o[nt][r] * inv);
    }
}

// ---------------------------------------------------------------------------
// Output projection: out = Y @ Wo^T + bo (fp32 result). Same async-LDS weight
// staging as qkv_gemm. grid = (1024/64, 8192/128).
// ---------------------------------------------------------------------------
__global__ __launch_bounds__(256) void out_gemm(
    const bf16* __restrict__ Y, const bf16* __restrict__ W,
    const float* __restrict__ bias, float* __restrict__ out) {

    __shared__ __align__(16) bf16 wtile[2][64 * 64];

    const int tid   = threadIdx.x;
    const int lane  = tid & 31;
    const int wv    = tid >> 5;
    const int mbase = blockIdx.y * 128 + wv * 16;
    const int nbase = blockIdx.x * 64;

    const int crow = tid >> 2;
    const int cel  = (tid & 3) * 16;
    const bf16* wsrc = W + (size_t)(nbase + crow) * Cq + cel;
    const unsigned ldsdst = lds_off(&wtile[0][0]) + (unsigned)(crow * 64 + cel) * 2;

    async_ld_b128(ldsdst,      wsrc);
    async_ld_b128(ldsdst + 16, wsrc + 8);

    v8f acc[4] = {v8f{}, v8f{}, v8f{}, v8f{}};
    const bf16* arow = Y + (size_t)(mbase + (lane & 15)) * Cq;

    for (int k0 = 0; k0 < Cq; k0 += 64) {
        const unsigned cur = (unsigned)(k0 >> 6) & 1u;
        wait_async0();
        __syncthreads();
        if (k0 + 64 < Cq) {
            const bf16* s = wsrc + k0 + 64;
            const unsigned d = ldsdst + (cur ^ 1u) * (64 * 64 * 2);
            async_ld_b128(d,      s);
            async_ld_b128(d + 16, s + 8);
            __builtin_prefetch(arow + k0 + 64, 0, 3);
        }
        const bf16* bt = &wtile[cur][0];
#pragma unroll
        for (int ks = 0; ks < 2; ++ks) {
            v16bf a = load_a(arow, k0 + ks * 32, lane);
            v16bf bfr[4];
#pragma unroll
            for (int nt = 0; nt < 4; ++nt)
                bfr[nt] = load_b(bt, 64, nt * 16 + (lane & 15), ks * 32, lane);
#pragma unroll
            for (int nt = 0; nt < 4; ++nt)
                acc[nt] = wmma_bf16(a, bfr[nt], acc[nt]);
        }
    }

#pragma unroll
    for (int nt = 0; nt < 4; ++nt) {
        const int col = nbase + nt * 16 + (lane & 15);
        const float bv = bias[col];
#pragma unroll
        for (int r = 0; r < 8; ++r) {
            const int row = mbase + r + (lane < 16 ? 0 : 8);
            out[(size_t)row * Cq + col] = acc[nt][r] + bv;
        }
    }
}

// ---------------------------------------------------------------------------
// Host launcher
// ---------------------------------------------------------------------------
extern "C" void kernel_launch(void* const* d_in, const int* in_sizes, int n_in,
                              void* d_out, int out_size, void* d_ws, size_t ws_size,
                              hipStream_t stream) {
    const float* x     = (const float*)d_in[0];   // [B,T,C]
    const int*   amask = (const int*)d_in[1];     // [B,T]
    const float* Wqkv  = (const float*)d_in[2];   // [3C,C]
    const float* bqkv  = (const float*)d_in[3];   // [3C]
    const float* Wo    = (const float*)d_in[4];   // [C,C]
    const float* bo    = (const float*)d_in[5];   // [C]
    float*       out   = (float*)d_out;           // [B,T,C]

    // workspace layout (bf16 staging), 256B-aligned sections
    char* ws = (char*)d_ws;
    bf16* xb    = (bf16*)(ws);                               // 16 MB
    bf16* wqkvb = (bf16*)(ws + (size_t)16 * 1024 * 1024);    //  6 MB
    bf16* wob   = (bf16*)(ws + (size_t)22 * 1024 * 1024);    //  2 MB
    bf16* Qb    = (bf16*)(ws + (size_t)24 * 1024 * 1024);    // 16 MB
    bf16* Kb    = (bf16*)(ws + (size_t)40 * 1024 * 1024);    // 16 MB
    bf16* Vtb   = (bf16*)(ws + (size_t)56 * 1024 * 1024);    // 16 MB
    bf16* Yb    = (bf16*)(ws + (size_t)72 * 1024 * 1024);    // 16 MB

    // 1) fp32 -> bf16 staging
    {
        int n = NTOK * Cq;          // 8388608
        cvt_f32_bf16<<<dim3((n / 4 + 255) / 256), dim3(256), 0, stream>>>(x, xb, n);
        n = C3q * Cq;               // 3145728
        cvt_f32_bf16<<<dim3((n / 4 + 255) / 256), dim3(256), 0, stream>>>(Wqkv, wqkvb, n);
        n = Cq * Cq;                // 1048576
        cvt_f32_bf16<<<dim3((n / 4 + 255) / 256), dim3(256), 0, stream>>>(Wo, wob, n);
    }

    // 2) QKV projection + head-major scatter (Q pre-scaled, V transposed)
    qkv_gemm<<<dim3(C3q / 64, NTOK / 128), dim3(256), 0, stream>>>(
        xb, wqkvb, bqkv, Qb, Kb, Vtb);

    // 3) flash attention per (b,h), 16 queries per wave
    attn_kernel<<<dim3(Tq / 128, Hq, Bq), dim3(256), 0, stream>>>(
        Qb, Kb, Vtb, amask, Yb);

    // 4) output projection -> fp32
    out_gemm<<<dim3(Cq / 64, NTOK / 128), dim3(256), 0, stream>>>(
        Yb, wob, bo, out);
}
